// AttFKANBlock_56977036148765
// MI455X (gfx1250) — compile-verified, matrix operand order checked by reading, not run on hardware
//
#include <hip/hip_runtime.h>
#include <hip/hip_bf16.h>

// ---------------------------------------------------------------------------
// AttFKANBlock for MI455X (gfx1250, wave32, WMMA, async global->LDS copies).
// FKAN layers recast as f16 GEMMs: [B=2048 x K=32768] x [K x O=2048].
// Block tile 128x256, wave tile 64x64 (16 WMMA per 8 fragments per k-step).
// ---------------------------------------------------------------------------

typedef __attribute__((ext_vector_type(16))) _Float16 v16h;
typedef __attribute__((ext_vector_type(8)))  float    v8f;

#define DIMC 2048
#define GR   8
#define IG   (DIMC * GR)     // 16384
#define KDIM (2 * IG)        // 32768  (cos block | sin block)
#define BATCH 2048
#define HID  (DIMC / 16)     // 128

#define TM 128
#define TN 256
#define KT 64
#define NTILES (KDIM / KT)   // 512 (even)

// ---------------------------------------------------------------------------
// CDNA5 async global->LDS copies (ASYNCcnt). INST_OFFSET applies to both the
// LDS and global address (ISA 08 Sec 4.4).
// ---------------------------------------------------------------------------
__device__ inline void async_copy64B(_Float16* lds_dst,
                                     const _Float16* gbase,
                                     unsigned gofs_bytes)
{
    unsigned lds = (unsigned)(unsigned long long)lds_dst;     // low 32b = LDS offset
    unsigned long long sb = (unsigned long long)gbase;
    asm volatile(
        "global_load_async_to_lds_b128 %0, %1, %2\n\t"
        "global_load_async_to_lds_b128 %0, %1, %2 offset:16\n\t"
        "global_load_async_to_lds_b128 %0, %1, %2 offset:32\n\t"
        "global_load_async_to_lds_b128 %0, %1, %2 offset:48"
        :: "v"(lds), "v"(gofs_bytes), "s"(sb)
        : "memory");
}

__device__ inline void async_copy128B(_Float16* lds_dst,
                                      const _Float16* gbase,
                                      unsigned gofs_bytes)
{
    unsigned lds = (unsigned)(unsigned long long)lds_dst;
    unsigned long long sb = (unsigned long long)gbase;
    asm volatile(
        "global_load_async_to_lds_b128 %0, %1, %2\n\t"
        "global_load_async_to_lds_b128 %0, %1, %2 offset:16\n\t"
        "global_load_async_to_lds_b128 %0, %1, %2 offset:32\n\t"
        "global_load_async_to_lds_b128 %0, %1, %2 offset:48\n\t"
        "global_load_async_to_lds_b128 %0, %1, %2 offset:64\n\t"
        "global_load_async_to_lds_b128 %0, %1, %2 offset:80\n\t"
        "global_load_async_to_lds_b128 %0, %1, %2 offset:96\n\t"
        "global_load_async_to_lds_b128 %0, %1, %2 offset:112"
        :: "v"(lds), "v"(gofs_bytes), "s"(sb)
        : "memory");
}

__device__ inline void wait_async0()
{
#if __has_builtin(__builtin_amdgcn_s_wait_asynccnt)
    __builtin_amdgcn_s_wait_asynccnt(0);
#else
    asm volatile("s_wait_asynccnt 0" ::: "memory");
#endif
}

// ---------------------------------------------------------------------------
// 1) Repack fp32 coef [2, O, I, G] -> f16 Wt [O, KDIM], Wt[o][t*IG + j] = c[t][o][j]
// ---------------------------------------------------------------------------
__global__ __launch_bounds__(256) void pack_weights(const float* __restrict__ src,
                                                    _Float16* __restrict__ dst)
{
    size_t idx = ((size_t)blockIdx.x * 256 + threadIdx.x) * 4;   // over 2*O*IG
    size_t t   = idx / ((size_t)DIMC * IG);
    size_t rem = idx - t * (size_t)DIMC * IG;
    size_t o   = rem / IG;
    size_t j   = rem - o * IG;
    const float4 v = *(const float4*)(src + idx);
    _Float16 h4[4] = { (_Float16)v.x, (_Float16)v.y, (_Float16)v.z, (_Float16)v.w };
    *(uint2*)(dst + o * KDIM + t * IG + j) = *(const uint2*)h4;
}

// ---------------------------------------------------------------------------
// 2) Layernorm + Fourier featurization: feat[b, i*8+g] = cos(v*(g+1)) (f16),
//    feat[b, IG + i*8+g] = sin(v*(g+1)), v = layernorm(x)[i]*w[i]+b[i]
// ---------------------------------------------------------------------------
__global__ __launch_bounds__(256) void featurize(const float* __restrict__ X,
                                                 const float* __restrict__ w,
                                                 const float* __restrict__ bias,
                                                 _Float16* __restrict__ feat)
{
    const int row = blockIdx.x;
    const int tid = threadIdx.x;
    const float* x = X + (size_t)row * DIMC;

    float xv[8];
    float s = 0.f, ss = 0.f;
#pragma unroll
    for (int j = 0; j < 8; ++j) {
        xv[j] = x[tid + j * 256];
        s  += xv[j];
        ss += xv[j] * xv[j];
    }
    __shared__ float r1[256], r2[256];
    r1[tid] = s; r2[tid] = ss;
    __syncthreads();
    for (int off = 128; off > 0; off >>= 1) {
        if (tid < off) { r1[tid] += r1[tid + off]; r2[tid] += r2[tid + off]; }
        __syncthreads();
    }
    const float mu  = r1[0] * (1.f / DIMC);
    const float var = r2[0] * (1.f / DIMC) - mu * mu;
    const float rs  = rsqrtf(var + 1e-5f);

    _Float16* fr = feat + (size_t)row * KDIM;
#pragma unroll
    for (int j = 0; j < 8; ++j) {
        const int i = tid + j * 256;
        const float v = (xv[j] - mu) * rs * w[i] + bias[i];
        _Float16 ch[8], sh[8];
#pragma unroll
        for (int g = 0; g < 8; ++g) {
            float sn, cs;
            __sincosf(v * (float)(g + 1), &sn, &cs);
            ch[g] = (_Float16)cs;
            sh[g] = (_Float16)sn;
        }
        *(uint4*)(fr + (size_t)i * 8)      = *(const uint4*)ch;
        *(uint4*)(fr + IG + (size_t)i * 8) = *(const uint4*)sh;
    }
}

// ---------------------------------------------------------------------------
// 3) f16 WMMA GEMM, double-buffered async pipeline.
//    Fragment load matches CDNA5 16-bit A layout:
//      lane L: row M = L%16; halfs[0..7] = K = 8*(L/16)+0..7; halfs[8..15] = +16
//    B (by output column, row-major in Wt) uses the identical pattern with N.
// ---------------------------------------------------------------------------
__device__ inline v16h ld_frag(const _Float16* lds, int row0, int lane, int ks)
{
    const int m  = lane & 15;
    const int hi = (lane >> 4) << 3;                 // 0 or 8
    const _Float16* p = lds + (row0 + m) * KT + ks + hi;
    union { v16h v; uint4 q[2]; } u;
    u.q[0] = *(const uint4*)(p);
    u.q[1] = *(const uint4*)(p + 16);
    return u.v;
}

__global__ __launch_bounds__(256) void gemm_f16(const _Float16* __restrict__ A,
                                                const _Float16* __restrict__ Wt,
                                                const float* __restrict__ bias,
                                                float* __restrict__ Y,
                                                int relu)
{
    __shared__ __align__(16) _Float16 sA[2][TM * KT];   // 2 x 16 KB
    __shared__ __align__(16) _Float16 sB[2][TN * KT];   // 2 x 32 KB

    const int tid  = threadIdx.x;
    const int lane = tid & 31;
    const int wave = tid >> 5;       // 0..7
    const int wm   = wave & 1;       // 2 waves along M -> 64 rows each
    const int wn   = wave >> 1;      // 4 waves along N -> 64 cols each
    const int bm   = blockIdx.y * TM;
    const int bn   = blockIdx.x * TN;

    v8f acc[4][4] = {};

    // A staging: thread t -> row t/2, 64B half-row. B staging: thread t -> row t (128B).
    const int srow  = tid >> 1;
    const int scolh = (tid & 1) * 32;
    const unsigned arow = (unsigned)(((unsigned)(bm + srow) * KDIM + scolh) * 2u);
    const unsigned brow = (unsigned)(((unsigned)(bn + tid) * KDIM) * 2u);

    auto stage = [&](int kt, int buf) {
        async_copy64B(&sA[buf][srow * KT + scolh], A,  arow + (unsigned)kt * 2u);
        async_copy128B(&sB[buf][tid * KT],         Wt, brow + (unsigned)kt * 2u);
    };

    auto compute = [&](const _Float16* bufA, const _Float16* bufB) {
#pragma unroll
        for (int ks = 0; ks < KT; ks += 32) {
            v16h bf[4];
#pragma unroll
            for (int j = 0; j < 4; ++j) bf[j] = ld_frag(bufB, wn * 64 + j * 16, lane, ks);
#pragma unroll
            for (int i = 0; i < 4; ++i) {
                const v16h af = ld_frag(bufA, wm * 64 + i * 16, lane, ks);
#pragma unroll
                for (int j = 0; j < 4; ++j)
                    acc[i][j] = __builtin_amdgcn_wmma_f32_16x16x32_f16(
                        false, af, false, bf[j], (short)0, acc[i][j], false, false);
            }
        }
    };

    // ---- software pipeline: prefetch t+1 (async) while computing t ----
    stage(0, 0);
    wait_async0();
    __syncthreads();
    for (int t = 0; t < NTILES; t += 2) {
        stage((t + 1) * KT, 1);            // t+1 <= NTILES-1 always
        compute(sA[0], sB[0]);
        wait_async0();
        __syncthreads();
        if (t + 2 < NTILES) stage((t + 2) * KT, 0);
        compute(sA[1], sB[1]);
        wait_async0();
        __syncthreads();
    }

    // ---- epilogue: C layout VGPR r -> M = r + 8*(lane>=16), N = lane&15 ----
    const int nlo = lane & 15;
    const int mhi = (lane >> 4) << 3;
#pragma unroll
    for (int i = 0; i < 4; ++i) {
        const int gm0 = bm + wm * 64 + i * 16 + mhi;
#pragma unroll
        for (int j = 0; j < 4; ++j) {
            const int gn = bn + wn * 64 + j * 16 + nlo;
            const float bv = bias[gn];
#pragma unroll
            for (int r = 0; r < 8; ++r) {
                float v = acc[i][j][r] + bv;
                if (relu) v = v > 0.f ? v : 0.f;
                Y[(size_t)(gm0 + r) * DIMC + gn] = v;
            }
        }
    }
}

// ---------------------------------------------------------------------------
// 4) LCBAM tail. Spatial map is 1x1 => avgpool == maxpool == the row, so
//    ca = sigmoid(2*mlp(y)); 7x7 conv on 1x1 keeps only center tap w[..,3,3].
// ---------------------------------------------------------------------------
__global__ __launch_bounds__(256) void tail(const float* __restrict__ X,
                                            const float* __restrict__ Y,
                                            const float* __restrict__ w1,
                                            const float* __restrict__ b1,
                                            const float* __restrict__ w2,
                                            const float* __restrict__ b2,
                                            const float* __restrict__ saw,
                                            const float* __restrict__ sab,
                                            float* __restrict__ Out)
{
    const int row = blockIdx.x;
    const int tid = threadIdx.x;
    __shared__ float sv[DIMC];
    __shared__ float sh[HID];
    __shared__ float sxc[DIMC];
    __shared__ float rsum[256], rmax[256];

    const float* y = Y + (size_t)row * DIMC;
#pragma unroll
    for (int j = 0; j < 8; ++j) sv[tid + j * 256] = y[tid + j * 256];
    __syncthreads();

    if (tid < HID) {
        float a = b1[tid];
        const float* wr = w1 + (size_t)tid * DIMC;
        for (int c = 0; c < DIMC; ++c) a += wr[c] * sv[c];
        sh[tid] = a > 0.f ? a : 0.f;
    }
    __syncthreads();

    float psum = 0.f, pmax = -3.4e38f;
#pragma unroll
    for (int j = 0; j < 8; ++j) {
        const int c = tid + j * 256;
        float a = b2[c];
        const float* wr = w2 + (size_t)c * HID;
        for (int k = 0; k < HID; ++k) a += wr[k] * sh[k];
        const float ca = 1.f / (1.f + __expf(-2.f * a));   // mlp(avg)+mlp(max) = 2*mlp(v)
        const float xc = sv[c] * ca;
        sxc[c] = xc;
        psum += xc;
        pmax = fmaxf(pmax, xc);
    }
    rsum[tid] = psum; rmax[tid] = pmax;
    __syncthreads();
    for (int off = 128; off > 0; off >>= 1) {
        if (tid < off) {
            rsum[tid] += rsum[tid + off];
            rmax[tid] = fmaxf(rmax[tid], rmax[tid + off]);
        }
        __syncthreads();
    }
    // sa_w[1,2,7,7]: center taps (0,0,3,3)=24, (0,1,3,3)=73
    const float sa = rsum[0] * (1.f / DIMC) * saw[24] + rmax[0] * saw[73] + sab[0];
    const float sg = 1.f / (1.f + __expf(-sa));

    const float* xr   = X + (size_t)row * DIMC;
    float*       outr = Out + (size_t)row * DIMC;
#pragma unroll
    for (int j = 0; j < 8; ++j) {
        const int c = tid + j * 256;
        outr[c] = xr[c] + sxc[c] * sg;
    }
}

// ---------------------------------------------------------------------------
extern "C" void kernel_launch(void* const* d_in, const int* in_sizes, int n_in,
                              void* d_out, int out_size, void* d_ws, size_t ws_size,
                              hipStream_t stream)
{
    (void)in_sizes; (void)n_in; (void)out_size; (void)ws_size;
    const float* x     = (const float*)d_in[0];
    const float* n1_w  = (const float*)d_in[1];
    const float* n1_b  = (const float*)d_in[2];
    const float* fk1_c = (const float*)d_in[3];
    const float* fk1_b = (const float*)d_in[4];
    const float* n2_w  = (const float*)d_in[5];
    const float* n2_b  = (const float*)d_in[6];
    const float* fk2_c = (const float*)d_in[7];
    const float* fk2_b = (const float*)d_in[8];
    const float* ca_w1 = (const float*)d_in[9];
    const float* ca_b1 = (const float*)d_in[10];
    const float* ca_w2 = (const float*)d_in[11];
    const float* ca_b2 = (const float*)d_in[12];
    const float* sa_w  = (const float*)d_in[13];
    const float* sa_b  = (const float*)d_in[14];

    const size_t WT_BYTES   = (size_t)DIMC * KDIM * sizeof(_Float16);   // 128 MB
    const size_t FEAT_BYTES = (size_t)BATCH * KDIM * sizeof(_Float16);  // 128 MB
    const size_t Y_BYTES    = (size_t)BATCH * DIMC * sizeof(float);     // 16 MB

    char* ws = (char*)d_ws;
    _Float16* Wt1  = (_Float16*)(ws);
    _Float16* Wt2  = (_Float16*)(ws + WT_BYTES);
    _Float16* feat = (_Float16*)(ws + 2 * WT_BYTES);
    float*    y1   = (float*)(ws + 2 * WT_BYTES + FEAT_BYTES);
    float*    y2   = (float*)(ws + 2 * WT_BYTES + FEAT_BYTES + Y_BYTES);

    const int packBlocks = (int)(((size_t)2 * DIMC * IG) / (4 * 256));  // 65536
    pack_weights<<<packBlocks, 256, 0, stream>>>(fk1_c, Wt1);
    pack_weights<<<packBlocks, 256, 0, stream>>>(fk2_c, Wt2);

    dim3 ggrid(DIMC / TN, BATCH / TM);   // 8 x 16

    featurize<<<BATCH, 256, 0, stream>>>(x, n1_w, n1_b, feat);
    gemm_f16<<<ggrid, 256, 0, stream>>>(feat, Wt1, fk1_b, y1, 1);
    featurize<<<BATCH, 256, 0, stream>>>(y1, n2_w, n2_b, feat);
    gemm_f16<<<ggrid, 256, 0, stream>>>(feat, Wt2, fk2_b, y2, 0);
    tail<<<BATCH, 256, 0, stream>>>(x, y2, ca_w1, ca_b1, ca_w2, ca_b2,
                                    sa_w, sa_b, (float*)d_out);
}
